// FFM_47665547051504
// MI455X (gfx1250) — matrix operand order repaired and checked by reading, not compiled, not run on hardware
//
#include <hip/hip_runtime.h>
#include <hip/hip_bf16.h>
#include <math.h>

// ---------------- problem dims ----------------
#define T_DIM   4096
#define IN_DIM  512
#define TR_DIM  128
#define CTX_DIM 32
#define OUT_DIM 512
#define NCH     (TR_DIM * CTX_DIM)      // 4096 scan channels
#define ZW      (2 * TR_DIM * CTX_DIM)  // 8192, width of z_in
#define LN_EPS  1e-6f

#define KSLAB     64                    // K-depth staged per LDS buffer
#define LDS_PITCH 72                    // 64 + 8 pad (bf16 elems) -> bank spread

typedef __attribute__((ext_vector_type(16))) __bf16 v16bf;
typedef __attribute__((ext_vector_type(8)))  __bf16 v8bf;
typedef __attribute__((ext_vector_type(8)))  float  v8f;
typedef __attribute__((ext_vector_type(4)))  int    v4i;

// async-to-LDS builtin expects int4 pointers in explicit address spaces
typedef __attribute__((address_space(1))) v4i gas_v4i;  // global
typedef __attribute__((address_space(3))) v4i las_v4i;  // LDS

// ---------------- CDNA5 async copy helpers ----------------
__device__ inline void async_ld16(const __bf16* g, __bf16* l) {
#if defined(__has_builtin) && __has_builtin(__builtin_amdgcn_global_load_async_to_lds_b128)
  // integer round-trip: generic LDS address low 32 bits == LDS offset
  __builtin_amdgcn_global_load_async_to_lds_b128(
      (gas_v4i*)(unsigned long long)g,
      (las_v4i*)(unsigned)(unsigned long long)l,
      0, 0);
#else
  asm volatile("global_load_async_to_lds_b128 %0, %1, off"
               :: "v"((unsigned)(unsigned long long)l), "v"(g)
               : "memory");
#endif
}

__device__ inline void wait_async0() {
#if defined(__has_builtin) && __has_builtin(__builtin_amdgcn_s_wait_asynccnt)
  __builtin_amdgcn_s_wait_asynccnt(0);
#else
  asm volatile("s_wait_asynccnt 0x0" ::: "memory");
#endif
}

// ---------------- WMMA helpers (gfx1250 wave32) ----------------
__device__ inline v16bf cat8(v8bf lo, v8bf hi) {
  return __builtin_shufflevector(lo, hi, 0,1,2,3,4,5,6,7,8,9,10,11,12,13,14,15);
}

// A fragment read out of the LDS slab (pitch LDS_PITCH), per ISA 7.12.2:
// lanes 0-15: M=lane, K = {kk..kk+7, kk+16..kk+23}; lanes 16-31: +8 on K.
__device__ inline v16bf lds_a_frag(const __bf16* base, int kk, int lane) {
  int half = lane >> 4;
  int m    = lane & 15;
  const __bf16* p = base + m * LDS_PITCH + kk + half * 8;
  v8bf lo = *(const v8bf*)(p);
  v8bf hi = *(const v8bf*)(p + 16);
  return cat8(lo, hi);
}

// B fragment: Bt is the transposed weight (N x K); lanes 0-15 hold
// K = {k0..k0+7, k0+16..k0+23} (N=lane), lanes 16-31: +8 on K.
__device__ inline v16bf load_b_frag(const __bf16* __restrict__ Bt, int K,
                                    int col0, int k0, int lane) {
  int half = lane >> 4;
  int n    = lane & 15;
  const __bf16* p = Bt + (size_t)(col0 + n) * K + k0 + half * 8;
  v8bf lo = *(const v8bf*)(p);
  v8bf hi = *(const v8bf*)(p + 16);
  return cat8(lo, hi);
}

__device__ inline v8f wmma_bf16(v16bf a, v16bf b, v8f c) {
  return __builtin_amdgcn_wmma_f32_16x16x32_bf16(
      /*neg_a=*/false, a, /*neg_b=*/false, b,
      /*c_mod=*/(short)0, c, /*reuse_a=*/false, /*reuse_b=*/false);
}

// ---------------- conversion / packing kernels ----------------
__global__ void k_f32_to_bf16(const float* __restrict__ s, __bf16* __restrict__ d, int n) {
  int i = blockIdx.x * blockDim.x + threadIdx.x;
  if (i < n) d[i] = (__bf16)s[i];
}

// W is (K x Nsrc) row-major f32; write transposed bf16 into Bt at column
// offset col_off: Bt[(col_off + n) * K + k] = W[k*Nsrc + n].
__global__ void k_pack_wt(const float* __restrict__ W, __bf16* __restrict__ Bt,
                          int K, int Nsrc, int col_off) {
  int i = blockIdx.x * blockDim.x + threadIdx.x;
  int n_elems = K * Nsrc;
  if (i >= n_elems) return;
  int k = i / Nsrc;
  int n = i % Nsrc;
  Bt[(size_t)(col_off + n) * K + k] = (__bf16)W[i];
}

__global__ void k_copy_f32(const float* __restrict__ s, float* __restrict__ d, int n) {
  int i = blockIdx.x * blockDim.x + threadIdx.x;
  if (i < n) d[i] = s[i];
}

// ---------------- GEMM: C(MxN) = A(MxK) * Bt(NxK)^T + bias ----------------
// block = 256 threads = 8 waves; each wave computes a 32x32 output tile.
// All 8 waves of a block share one 32-row A slab (tiles_n is a multiple of
// 8 in every call), so A is staged through LDS with async-to-LDS double
// buffering; B streams straight from global (distinct columns per wave).
__global__ void k_gemm_bf16(const __bf16* __restrict__ A,
                            const __bf16* __restrict__ Bt,
                            const float*  __restrict__ bias,
                            float* __restrict__ C,
                            int M, int N, int K) {
  __shared__ __bf16 sA[2][32 * LDS_PITCH];

  int tid  = threadIdx.x;
  int lane = tid & 31;
  int wave = tid >> 5;
  int tile = blockIdx.x * 8 + wave;
  int tiles_n = N >> 5;
  int tm = tile / tiles_n;
  int tn = tile - tm * tiles_n;
  int row0 = tm << 5;            // identical for all 8 waves of the block
  int col0 = tn << 5;

  // per-thread async-copy assignment: 16 bytes each, 32 rows x 64 K
  int ldrow = tid >> 3;          // 0..31
  int ldkq  = (tid & 7) * 8;     // 0,8,...,56
  const __bf16* gA = A + (size_t)(row0 + ldrow) * K + ldkq;
  __bf16* lA0 = &sA[0][ldrow * LDS_PITCH + ldkq];
  __bf16* lA1 = &sA[1][ldrow * LDS_PITCH + ldkq];

  v8f c00 = {}, c01 = {}, c10 = {}, c11 = {};

  // prologue: stage slab 0
  async_ld16(gA, lA0);
  wait_async0();
  __syncthreads();

  int nslab = K / KSLAB;
  for (int s = 0; s < nslab; ++s) {
    const __bf16* cur = sA[s & 1];
    if (s + 1 < nslab) {
      async_ld16(gA + (s + 1) * KSLAB, (s & 1) ? lA0 : lA1);
      __builtin_prefetch(Bt + (size_t)(col0 + (lane & 15)) * K + (s + 1) * KSLAB, 0, 1);
    }
    int kbase = s * KSLAB;
#pragma unroll
    for (int kk = 0; kk < KSLAB; kk += 32) {
      v16bf a0 = lds_a_frag(cur,                  kk, lane);  // rows 0-15
      v16bf a1 = lds_a_frag(cur + 16 * LDS_PITCH, kk, lane);  // rows 16-31
      v16bf b0 = load_b_frag(Bt, K, col0,      kbase + kk, lane);
      v16bf b1 = load_b_frag(Bt, K, col0 + 16, kbase + kk, lane);
      c00 = wmma_bf16(a0, b0, c00);
      c01 = wmma_bf16(a0, b1, c01);
      c10 = wmma_bf16(a1, b0, c10);
      c11 = wmma_bf16(a1, b1, c11);
    }
    wait_async0();   // loads into the other buffer are complete
    __syncthreads(); // publish to all waves / allow overwrite next iter
  }

  // C/D layout: VGPR r holds M = row0 + r + 8*half, N = col0 + (lane&15)
  int n    = lane & 15;
  int half = lane >> 4;
  float bn0 = bias[col0 + n];
  float bn1 = bias[col0 + 16 + n];
#pragma unroll
  for (int r = 0; r < 8; ++r) {
    int m = row0 + r + 8 * half;
    C[(size_t)m * N + col0 + n]             = c00[r] + bn0;
    C[(size_t)m * N + col0 + 16 + n]        = c01[r] + bn1;
    C[(size_t)(m + 16) * N + col0 + n]      = c10[r] + bn0;
    C[(size_t)(m + 16) * N + col0 + 16 + n] = c11[r] + bn1;
  }
}

// ---------------- gated = pre * sigmoid(gate_in) ----------------
// C1 is (T x 256): cols [0,128) = pre, [128,256) = gate_in
__global__ void k_gated(const float* __restrict__ C1, float* __restrict__ gated, int n) {
  int i = blockIdx.x * blockDim.x + threadIdx.x;
  if (i >= n) return;
  int t  = i / TR_DIM;
  int tr = i - t * TR_DIM;
  float pre = C1[(size_t)t * 256 + tr];
  float g   = C1[(size_t)t * 256 + 128 + tr];
  gated[i] = pre * (1.0f / (1.0f + expf(-g)));
}

// ---------------- complex linear-recurrence scan ----------------
// One thread per (tr,ctx) channel. s_t = lambda * s_{t-1} + gated[t,tr].
// Emits z bf16 (T x 8192): z[t][tr*64 + c] = Re, z[t][tr*64 + 32 + c] = Im.
// Also writes final state + t_all into the output tail.
__global__ void k_scan(const float* __restrict__ gated,
                       const float* __restrict__ ffa_a,
                       const float* __restrict__ ffa_b,
                       const float* __restrict__ state0, // complex64 interleaved
                       const int*   __restrict__ ts0,
                       __bf16* __restrict__ z,
                       float* __restrict__ out_tail) {
  int ch = blockIdx.x * blockDim.x + threadIdx.x;
  if (ch >= NCH) return;
  int tr = ch >> 5;        // /32
  int c  = ch & 31;

  float er = expf(-fabsf(ffa_a[tr]));
  float b  = ffa_b[c];
  float lr = er * cosf(b);
  float li = er * sinf(b);

  float sr = state0[2 * ch + 0];
  float si = state0[2 * ch + 1];

  size_t zoff = (size_t)tr * 64 + c;
  for (int t = 0; t < T_DIM; ++t) {
    float g  = gated[(size_t)t * TR_DIM + tr];
    float nr = lr * sr - li * si + g;
    float ni = lr * si + li * sr;
    sr = nr; si = ni;
    __bf16* zp = z + (size_t)t * ZW + zoff;
    zp[0]  = (__bf16)sr;
    zp[32] = (__bf16)si;
  }

  out_tail[ch]        = sr;   // real(final_state)
  out_tail[NCH + ch]  = si;   // imag(final_state)
  if (ch == 0) {
    ((int*)out_tail)[2 * NCH] = ts0[0] + T_DIM;  // t_all[-1]
  }
}

// ---------------- final: gate/skip + LayerNorm + residual ----------------
// C3 is (T x 1024): cols [0,512) = gate_out pre-sigmoid, [512,1024) = skip.
__global__ void k_final(const float* __restrict__ zmix,
                        const float* __restrict__ C3,
                        float* __restrict__ out) {
  __shared__ float ssum[8];
  __shared__ float ssq[8];
  __shared__ float stat[2];

  int t    = blockIdx.x;
  int tid  = threadIdx.x;       // 256 threads, 2 cols each
  int lane = tid & 31;
  int wave = tid >> 5;

  float h[2], sk[2], go[2];
  float psum = 0.f, psq = 0.f;
#pragma unroll
  for (int e = 0; e < 2; ++e) {
    int j = tid + e * 256;
    float g = C3[(size_t)t * 1024 + j];
    go[e] = 1.0f / (1.0f + expf(-g));
    sk[e] = C3[(size_t)t * 1024 + 512 + j];
    float hv = zmix[(size_t)t * OUT_DIM + j] * go[e];
    h[e] = hv;
    psum += hv;
    psq  += hv * hv;
  }
  for (int off = 16; off > 0; off >>= 1) {
    psum += __shfl_down(psum, off);
    psq  += __shfl_down(psq, off);
  }
  if (lane == 0) { ssum[wave] = psum; ssq[wave] = psq; }
  __syncthreads();
  if (tid == 0) {
    float a = 0.f, b = 0.f;
    for (int w = 0; w < 8; ++w) { a += ssum[w]; b += ssq[w]; }
    float mu  = a / OUT_DIM;
    float var = b / OUT_DIM - mu * mu;
    stat[0] = mu;
    stat[1] = rsqrtf(var + LN_EPS);
  }
  __syncthreads();
  float mu = stat[0], rs = stat[1];
#pragma unroll
  for (int e = 0; e < 2; ++e) {
    int j = tid + e * 256;
    out[(size_t)t * OUT_DIM + j] = (h[e] - mu) * rs + sk[e] * (1.0f - go[e]);
  }
}

// ---------------- host launch ----------------
static inline int cdiv(int a, int b) { return (a + b - 1) / b; }

extern "C" void kernel_launch(void* const* d_in, const int* in_sizes, int n_in,
                              void* d_out, int out_size, void* d_ws, size_t ws_size,
                              hipStream_t stream) {
  const float* x          = (const float*)d_in[0];
  // d_in[1] resets: all false in setup, unused
  const float* W_pre      = (const float*)d_in[2];
  const float* b_pre      = (const float*)d_in[3];
  const float* W_gate_in  = (const float*)d_in[4];
  const float* b_gate_in  = (const float*)d_in[5];
  const float* ffa_a      = (const float*)d_in[6];
  const float* ffa_b      = (const float*)d_in[7];
  const float* W_mix      = (const float*)d_in[8];
  const float* b_mix      = (const float*)d_in[9];
  const float* W_gate_out = (const float*)d_in[10];
  const float* b_gate_out = (const float*)d_in[11];
  const float* W_skip     = (const float*)d_in[12];
  const float* b_skip     = (const float*)d_in[13];
  const float* state0     = (const float*)d_in[14]; // complex64 interleaved
  const int*   ts0        = (const int*)d_in[15];
  float* out = (float*)d_out;

  // workspace carve-up (all 256B aligned)
  char* ws = (char*)d_ws;
  size_t o = 0;
  __bf16* xb    = (__bf16*)(ws + o); o += (size_t)T_DIM * IN_DIM * 2;        // 4 MB
  __bf16* bt1   = (__bf16*)(ws + o); o += (size_t)256 * IN_DIM * 2;          // 256 KB
  __bf16* btm   = (__bf16*)(ws + o); o += (size_t)OUT_DIM * ZW * 2;          // 8 MB
  __bf16* bt3   = (__bf16*)(ws + o); o += (size_t)1024 * IN_DIM * 2;         // 1 MB
  float*  bias1 = (float*)(ws + o);  o += 256 * 4 + 768;                     // 1 KB + 768 B pad
  float*  bias3 = (float*)(ws + o);  o += 1024 * 4;                          // 4 KB
  float*  c1    = (float*)(ws + o);  o += (size_t)T_DIM * 256 * 4;           // 4 MB
  float*  gated = (float*)(ws + o);  o += (size_t)T_DIM * TR_DIM * 4;        // 2 MB
  __bf16* zb    = (__bf16*)(ws + o); o += (size_t)T_DIM * ZW * 2;            // 64 MB
  float*  zmix  = (float*)(ws + o);  o += (size_t)T_DIM * OUT_DIM * 4;       // 8 MB
  float*  c3    = (float*)(ws + o);  o += (size_t)T_DIM * 1024 * 4;          // 16 MB
  (void)ws_size; (void)in_sizes; (void)n_in; (void)out_size;

  const int B = 256;

  // --- conversions / packing ---
  k_f32_to_bf16<<<cdiv(T_DIM * IN_DIM, B), B, 0, stream>>>(x, xb, T_DIM * IN_DIM);
  k_pack_wt<<<cdiv(IN_DIM * TR_DIM, B), B, 0, stream>>>(W_pre,     bt1, IN_DIM, TR_DIM, 0);
  k_pack_wt<<<cdiv(IN_DIM * TR_DIM, B), B, 0, stream>>>(W_gate_in, bt1, IN_DIM, TR_DIM, TR_DIM);
  k_pack_wt<<<cdiv(ZW * OUT_DIM, B), B, 0, stream>>>(W_mix, btm, ZW, OUT_DIM, 0);
  k_pack_wt<<<cdiv(IN_DIM * OUT_DIM, B), B, 0, stream>>>(W_gate_out, bt3, IN_DIM, OUT_DIM, 0);
  k_pack_wt<<<cdiv(IN_DIM * OUT_DIM, B), B, 0, stream>>>(W_skip,     bt3, IN_DIM, OUT_DIM, OUT_DIM);
  k_copy_f32<<<1, B, 0, stream>>>(b_pre,     bias1,       TR_DIM);
  k_copy_f32<<<1, B, 0, stream>>>(b_gate_in, bias1 + 128, TR_DIM);
  k_copy_f32<<<cdiv(OUT_DIM, B), B, 0, stream>>>(b_gate_out, bias3,       OUT_DIM);
  k_copy_f32<<<cdiv(OUT_DIM, B), B, 0, stream>>>(b_skip,     bias3 + 512, OUT_DIM);

  // --- GEMM1: (4096x512)x(512x256) -> pre | gate_in ---
  {
    int tiles = (T_DIM / 32) * (256 / 32);   // 1024
    k_gemm_bf16<<<tiles / 8, B, 0, stream>>>(xb, bt1, bias1, c1, T_DIM, 256, IN_DIM);
  }
  k_gated<<<cdiv(T_DIM * TR_DIM, B), B, 0, stream>>>(c1, gated, T_DIM * TR_DIM);

  // --- scan: 4096 channels, writes z (bf16) + final state + t_all ---
  k_scan<<<NCH / B, B, 0, stream>>>(gated, ffa_a, ffa_b, state0, ts0, zb,
                                    out + (size_t)T_DIM * OUT_DIM);

  // --- GEMM2 (dominant): (4096x8192)x(8192x512) + b_mix ---
  {
    int tiles = (T_DIM / 32) * (OUT_DIM / 32); // 2048
    k_gemm_bf16<<<tiles / 8, B, 0, stream>>>(zb, btm, b_mix, zmix, T_DIM, OUT_DIM, ZW);
  }

  // --- GEMM3: (4096x512)x(512x1024) -> gate_out | skip ---
  {
    int tiles = (T_DIM / 32) * (1024 / 32);    // 4096
    k_gemm_bf16<<<tiles / 8, B, 0, stream>>>(xb, bt3, bias3, c3, T_DIM, 1024, IN_DIM);
  }

  // --- final gate + LayerNorm + residual ---
  k_final<<<T_DIM, B, 0, stream>>>(zmix, c3, out);
}
